// GNN_14860586844627
// MI455X (gfx1250) — matrix operand order repaired and checked by reading, not compiled, not run on hardware
//
#include <hip/hip_runtime.h>
#include <cstdint>
#include <cstddef>

// ---------- types ----------
typedef __attribute__((ext_vector_type(16))) _Float16 v16h;
typedef __attribute__((ext_vector_type(8)))  _Float16 v8h;
typedef __attribute__((ext_vector_type(8)))  float    v8f;

#define N_NODE 255
#define NPAD   256
#define NPAIR  65536      // NPAD*NPAD padded pair count
#define CDIM   256
#define J17C   4352       // 17*256

__device__ __forceinline__ float sigmoidf_(float x) { return 1.f / (1.f + __expf(-x)); }

// =====================================================================
// Generic WMMA GEMM:  Out[M,N] = act( A[M,K](f16,row-major) * B(from Bt[N,K] f16)
//                                     + bias[N] (+ P[i,col] + Q[j,col] if PQ) )
// One wave computes a 16x64 tile (4 col-tiles), block = 4 waves (128 thr).
// K multiple of 32, M multiple of 64, N multiple of 64.
// =====================================================================
template<bool RELU, bool OUTF16, bool PQ>
__global__ void k_gemm(const _Float16* __restrict__ A,
                       const _Float16* __restrict__ Bt,
                       const float* __restrict__ bias,
                       const float* __restrict__ Pm,
                       const float* __restrict__ Qm,
                       void* __restrict__ Out,
                       int M, int K, int N) {
  const int lane = threadIdx.x & 31;
  const int wave = threadIdx.x >> 5;
  const int rowBase = (blockIdx.y * 4 + wave) * 16;
  const int colBase = blockIdx.x * 64;
  if (rowBase >= M) return;

  const int m   = lane & 15;
  const int khi = (lane & 16) ? 16 : 0;   // A-fragment k offset (contiguous k = 16h + s, matches B)
  const int klo = (lane & 16) ? 16 : 0;   // B-fragment k offset (contiguous k = 16h + s, matches A)

  v8f acc[4];
#pragma unroll
  for (int t = 0; t < 4; ++t)
#pragma unroll
    for (int e = 0; e < 8; ++e) acc[t][e] = 0.f;

  const _Float16* Arow = A + (size_t)(rowBase + m) * K;

  for (int kc = 0; kc < K; kc += 32) {
    // A fragment: two contiguous 8-half (16B) vector loads
    v8h alo = *(const v8h*)(Arow + kc + khi);
    v8h ahi = *(const v8h*)(Arow + kc + 8 + khi);
    v16h a;
#pragma unroll
    for (int e = 0; e < 8; ++e) { a[e] = alo[e]; a[e + 8] = ahi[e]; }

#pragma unroll
    for (int t = 0; t < 4; ++t) {
      // B fragment: 16 contiguous halves (32B) from column-major weights
      v16h b = *(const v16h*)(Bt + (size_t)(colBase + t * 16 + m) * K + kc + klo);
      acc[t] = __builtin_amdgcn_wmma_f32_16x16x32_f16(
          false, a, false, b, (short)0, acc[t], false, false);
    }
  }

  // epilogue: C/D layout -> row = rowBase + r + 8*(lane>=16), col = colBase + 16t + (lane&15)
  const int rowOff = (lane & 16) ? 8 : 0;
#pragma unroll
  for (int t = 0; t < 4; ++t) {
    const int col = colBase + t * 16 + m;
    const float bval = bias ? bias[col] : 0.f;
#pragma unroll
    for (int r = 0; r < 8; ++r) {
      const int row = rowBase + rowOff + r;
      float v = acc[t][r] + bval;
      if (PQ) v += Pm[(size_t)(row >> 8) * CDIM + col] + Qm[(size_t)(row & 255) * CDIM + col];
      if (RELU) v = fmaxf(v, 0.f);
      if (OUTF16) ((_Float16*)Out)[(size_t)row * N + col] = (_Float16)v;
      else        ((float*)Out)[(size_t)row * N + col] = v;
    }
  }
}

// ---------- weight convert: f32 [Kin x Nin] row-major -> f16 [Nin x Kpad] (transposed, zero-pad K)
__global__ void k_wconv(const float* __restrict__ W, _Float16* __restrict__ Wt,
                        int Kin, int Kpad, int Nin) {
  size_t idx = (size_t)blockIdx.x * 256 + threadIdx.x;
  if (idx >= (size_t)Nin * Kpad) return;
  int n = (int)(idx / Kpad);
  int k = (int)(idx % Kpad);
  float v = (k < Kin) ? W[(size_t)k * Nin + n] : 0.f;
  Wt[idx] = (_Float16)v;
}

// ---------- node embed: nf[i] = nv[i] @ Wvf[jt[i]-block] + bvf,  nv = x@Wv + bv ----------
__global__ void k_node_embed(const float* __restrict__ vis, const float* __restrict__ Wv,
                             const float* __restrict__ bv, const float* __restrict__ Wvf,
                             const float* __restrict__ bvf, const int* __restrict__ jt,
                             float* __restrict__ nf) {
  int i = blockIdx.x, c = threadIdx.x;
  if (i >= N_NODE) { nf[(size_t)i * CDIM + c] = 0.f; return; }
  __shared__ float xr[CDIM], nvr[CDIM];
  xr[c] = vis[(size_t)i * CDIM + c];
  __syncthreads();
  float acc = bv[c];
  for (int k = 0; k < CDIM; ++k) acc += xr[k] * Wv[(size_t)k * CDIM + c];
  nvr[c] = acc;
  __syncthreads();
  const float* Wb = Wvf + (size_t)jt[i] * CDIM * CDIM;
  float a2 = bvf[c];
  for (int k = 0; k < CDIM; ++k) a2 += nvr[k] * Wb[(size_t)k * CDIM + c];
  nf[(size_t)i * CDIM + c] = a2;
}

// ---------- pairwise coordinate differences, padded K=64 ----------
__global__ void k_build_xa(const float* __restrict__ xy, _Float16* __restrict__ Axy) {
  size_t idx = (size_t)blockIdx.x * 256 + threadIdx.x;  // NPAIR*64 total
  int p = (int)(idx >> 6), k = (int)(idx & 63);
  int i = p >> 8, j = p & 255;
  float v = 0.f;
  if (i < N_NODE && j < N_NODE && k < 34) v = xy[i * 34 + k] - xy[j * 34 + k];
  Axy[idx] = (_Float16)v;
}

// ---------- |nf_i - nf_j| ----------
__global__ void k_build_da(const float* __restrict__ nf, _Float16* __restrict__ D) {
  int p = blockIdx.x, k = threadIdx.x;
  int i = p >> 8, j = p & 255;
  float v = fabsf(nf[(size_t)i * CDIM + k] - nf[(size_t)j * CDIM + k]);
  D[(size_t)p * CDIM + k] = (_Float16)v;
}

// ---------- last (K=64 -> 1) layer of an edge MLP ----------
__global__ void k_mlp_last(const _Float16* __restrict__ H, const float* __restrict__ w,
                           const float* __restrict__ b, float* __restrict__ out) {
  __shared__ float ws_[64];
  if (threadIdx.x < 64) ws_[threadIdx.x] = w[threadIdx.x];
  __syncthreads();
  int p = blockIdx.x * 256 + threadIdx.x;
  const _Float16* h = H + (size_t)p * 64;
  float acc = b[0];
#pragma unroll
  for (int k = 0; k < 64; ++k) acc += (float)h[k] * ws_[k];
  out[p] = acc;
}

// ---------- visual node update after iter 0 ----------
__global__ void k_vis_update(const float* __restrict__ evis, const float* __restrict__ nf,
                             const float* __restrict__ vWn, const float* __restrict__ vbn,
                             float* __restrict__ nfo) {
  int i = blockIdx.x, c = threadIdx.x;
  if (i >= N_NODE) { nfo[(size_t)i * CDIM + c] = 0.f; return; }
  __shared__ float sm[NPAD], agg[CDIM], nfr[CDIM];
  __shared__ float tot;
  float sv = 0.f;
  if (c < N_NODE && c != i) sv = sigmoidf_(evis[(size_t)i * NPAD + c]);
  sm[c] = sv;
  __syncthreads();
  if (c == 0) {
    float t = 0.f;
    for (int j = 0; j < N_NODE; ++j) t += sm[j];
    tot = fmaxf(t, 1e-12f);
  }
  __syncthreads();
  float a = 0.f;
  for (int j = 0; j < N_NODE; ++j) a += sm[j] * nf[(size_t)j * CDIM + c];
  a /= tot;
  agg[c] = a;
  nfr[c] = nf[(size_t)i * CDIM + c];
  __syncthreads();
  float o = vbn[c];
  for (int k = 0; k < CDIM; ++k) o += agg[k] * vWn[(size_t)k * CDIM + c];
  for (int k = 0; k < CDIM; ++k) o += nfr[k] * vWn[(size_t)(CDIM + k) * CDIM + c];
  nfo[(size_t)i * CDIM + c] = fmaxf(o, 0.f);
}

// ---------- eg[i,j] = v^2 / (max over joint-block of column + 1e-8), v optionally sigmoid ----------
__global__ void k_norm_joint(const float* __restrict__ src, int ldsrc, int sig,
                             float* __restrict__ eg) {
  int i = blockIdx.x, j = threadIdx.x;
  __shared__ float row[NPAD];
  __shared__ float gm[17];
  float v = 0.f;
  if (j < N_NODE) {
    v = src[(size_t)i * ldsrc + j];
    if (sig) v = sigmoidf_(v);
  }
  row[j] = v;
  __syncthreads();
  if (j < 17) {
    float mx = row[j * 15];
    for (int q = 1; q < 15; ++q) mx = fmaxf(mx, row[j * 15 + q]);
    gm[j] = mx;
  }
  __syncthreads();
  eg[(size_t)i * NPAD + j] = (j < N_NODE) ? v * v / (gm[j / 15] + 1e-8f) : 0.f;
}

// ---------- per-joint weighted mean pooling: normf[i, t*256+c] ----------
__global__ void k_segmean(const float* __restrict__ eg, const _Float16* __restrict__ ef,
                          _Float16* __restrict__ normf) {
  int i = blockIdx.x, t = blockIdx.y, c = threadIdx.x;
  size_t oidx = (size_t)i * J17C + (size_t)t * CDIM + c;
  if (i >= N_NODE) { normf[oidx] = (_Float16)0.f; return; }
  float acc = 0.f;
  int jb = t * 15;
  for (int q = 0; q < 15; ++q) {
    float w = eg[(size_t)i * NPAD + jb + q];
    acc += w * (float)ef[((size_t)(i * NPAD + jb + q)) * CDIM + c];
  }
  normf[oidx] = (_Float16)(acc * (1.0f / 15.0f));
}

// ---------- nfe[i] = relu(gbn + aggf[i] @ gWn[jt[i]-block]), f16 out for WMMA P/Q GEMMs ----------
__global__ void k_nfe(const float* __restrict__ aggf, const float* __restrict__ gWn_i,
                      const float* __restrict__ gbn_i, const int* __restrict__ jt,
                      _Float16* __restrict__ nfeh) {
  int i = blockIdx.x, c = threadIdx.x;
  if (i >= N_NODE) { nfeh[(size_t)i * CDIM + c] = (_Float16)0.f; return; }
  __shared__ float ar[CDIM];
  ar[c] = aggf[(size_t)i * CDIM + c];
  __syncthreads();
  const float* Wb = gWn_i + (size_t)jt[i] * CDIM * CDIM;
  float o = gbn_i[c];
  for (int k = 0; k < CDIM; ++k) o += ar[k] * Wb[(size_t)k * CDIM + c];
  nfeh[(size_t)i * CDIM + c] = (_Float16)fmaxf(o, 0.f);
}

// ---------- merge MLP 2->16->64->64->16->1, weights staged in LDS ----------
__global__ void k_merge(const float* __restrict__ sim, const float* __restrict__ evis,
                        const float* __restrict__ mW1, const float* __restrict__ mb1,
                        const float* __restrict__ mW2, const float* __restrict__ mb2,
                        const float* __restrict__ mW3, const float* __restrict__ mb3,
                        const float* __restrict__ mW4, const float* __restrict__ mb4,
                        const float* __restrict__ mW5, const float* __restrict__ mb5,
                        float* __restrict__ out) {
  __shared__ float w1[32], b1[16], w2[1024], b2[64], w3[4096], b3[64],
                   w4[1024], b4[16], w5[16], b5v;
  int tid = threadIdx.x;
  for (int x = tid; x < 32; x += 256) w1[x] = mW1[x];
  if (tid < 16) b1[tid] = mb1[tid];
  for (int x = tid; x < 1024; x += 256) w2[x] = mW2[x];
  if (tid < 64) b2[tid] = mb2[tid];
  for (int x = tid; x < 4096; x += 256) w3[x] = mW3[x];
  if (tid < 64) b3[tid] = mb3[tid];
  for (int x = tid; x < 1024; x += 256) w4[x] = mW4[x];
  if (tid < 16) b4[tid] = mb4[tid];
  if (tid < 16) w5[tid] = mW5[tid];
  if (tid == 0) b5v = mb5[0];
  __syncthreads();
  int i = blockIdx.x, j = tid;
  if (j >= N_NODE) return;
  float s = sim[(size_t)i * NPAD + j];
  float e = evis[(size_t)i * NPAD + j];
  float h1[16], h2[64], h3[64], h4[16];
#pragma unroll
  for (int c = 0; c < 16; ++c) h1[c] = fmaxf(s * w1[c] + e * w1[16 + c] + b1[c], 0.f);
  for (int c = 0; c < 64; ++c) {
    float a = b2[c];
#pragma unroll
    for (int k = 0; k < 16; ++k) a += h1[k] * w2[k * 64 + c];
    h2[c] = fmaxf(a, 0.f);
  }
  for (int c = 0; c < 64; ++c) {
    float a = b3[c];
    for (int k = 0; k < 64; ++k) a += h2[k] * w3[k * 64 + c];
    h3[c] = fmaxf(a, 0.f);
  }
  for (int c = 0; c < 16; ++c) {
    float a = b4[c];
    for (int k = 0; k < 64; ++k) a += h3[k] * w4[k * 16 + c];
    h4[c] = fmaxf(a, 0.f);
  }
  float o = b5v;
#pragma unroll
  for (int k = 0; k < 16; ++k) o += h4[k] * w5[k];
  out[(size_t)i * N_NODE + j] = o;
}

// =====================================================================
extern "C" void kernel_launch(void* const* d_in, const int* in_sizes, int n_in,
                              void* d_out, int out_size, void* d_ws, size_t ws_size,
                              hipStream_t stream) {
  (void)in_sizes; (void)n_in; (void)out_size; (void)ws_size;
  const float* xy   = (const float*)d_in[0];
  const float* vis  = (const float*)d_in[1];
  const float* edge = (const float*)d_in[2];
  const float* Wv   = (const float*)d_in[3];
  const float* bv   = (const float*)d_in[4];
  const float* Wvf  = (const float*)d_in[5];
  const float* bvf  = (const float*)d_in[6];
  const float* We1  = (const float*)d_in[7];
  const float* be1  = (const float*)d_in[8];
  const float* We2  = (const float*)d_in[9];
  const float* be2  = (const float*)d_in[10];
  const float* gW1  = (const float*)d_in[11];
  const float* gb1  = (const float*)d_in[12];
  const float* gW2  = (const float*)d_in[13];
  const float* gb2  = (const float*)d_in[14];
  const float* gWn  = (const float*)d_in[15];
  const float* gbn  = (const float*)d_in[16];
  const float* gA1  = (const float*)d_in[17];
  const float* gbA1 = (const float*)d_in[18];
  const float* gA2  = (const float*)d_in[19];
  const float* gbA2 = (const float*)d_in[20];
  const float* gA3  = (const float*)d_in[21];
  const float* gbA3 = (const float*)d_in[22];
  const float* vA1  = (const float*)d_in[23];
  const float* vbA1 = (const float*)d_in[24];
  const float* vA2  = (const float*)d_in[25];
  const float* vbA2 = (const float*)d_in[26];
  const float* vA3  = (const float*)d_in[27];
  const float* vbA3 = (const float*)d_in[28];
  const float* vWn  = (const float*)d_in[29];
  const float* vbn  = (const float*)d_in[30];
  const float* mW1  = (const float*)d_in[31];
  const float* mb1  = (const float*)d_in[32];
  const float* mW2  = (const float*)d_in[33];
  const float* mb2  = (const float*)d_in[34];
  const float* mW3  = (const float*)d_in[35];
  const float* mb3  = (const float*)d_in[36];
  const float* mW4  = (const float*)d_in[37];
  const float* mb4  = (const float*)d_in[38];
  const float* mW5  = (const float*)d_in[39];
  const float* mb5  = (const float*)d_in[40];
  const int*   jt   = (const int*)d_in[41];

  // ---- workspace layout ----
  char* base = (char*)d_ws;
  size_t off = 0;
  auto alloc = [&](size_t bytes) -> char* {
    char* p = base + off;
    off += (bytes + 255) & ~(size_t)255;
    return p;
  };
  float*    nf    = (float*)alloc((size_t)NPAD * CDIM * 4);
  float*    nf2   = (float*)alloc((size_t)NPAD * CDIM * 4);
  _Float16* nfeh  = (_Float16*)alloc((size_t)NPAD * CDIM * 2);
  float*    aggf  = (float*)alloc((size_t)NPAD * CDIM * 4);
  float*    Pbuf  = (float*)alloc((size_t)NPAD * CDIM * 4);
  float*    Qbuf  = (float*)alloc((size_t)NPAD * CDIM * 4);
  float*    eg    = (float*)alloc((size_t)NPAD * NPAD * 4);
  float*    evis  = (float*)alloc((size_t)NPAIR * 4);
  float*    sim   = (float*)alloc((size_t)NPAIR * 4);
  _Float16* normf = (_Float16*)alloc((size_t)NPAD * J17C * 2);
  _Float16* t1    = (_Float16*)alloc((size_t)NPAD * 1024 * 2);
  _Float16* B64   = (_Float16*)alloc((size_t)NPAIR * 64 * 2);   // Axy, later h2/g2
  _Float16* BIGA  = (_Float16*)alloc((size_t)NPAIR * CDIM * 2); // h1 / h
  _Float16* BIGB  = (_Float16*)alloc((size_t)NPAIR * CDIM * 2); // ef (persists)
  _Float16* BIGC  = (_Float16*)alloc((size_t)NPAIR * CDIM * 2); // d / g1
  _Float16* We1T  = (_Float16*)alloc(256 * 64 * 2);
  _Float16* We2T  = (_Float16*)alloc(256 * 256 * 2);
  _Float16 *vA1T[2], *vA2T[2], *gA1loT[2], *gA1midT[2], *gA1hiT[2], *gA2T[2], *gW1T[2], *gW2T[2];
  for (int i = 0; i < 2; ++i) {
    vA1T[i]   = (_Float16*)alloc(256 * 256 * 2);
    vA2T[i]   = (_Float16*)alloc(64 * 256 * 2);
    gA1loT[i] = (_Float16*)alloc(256 * 256 * 2);
    gA1midT[i]= (_Float16*)alloc(256 * 256 * 2);
    gA1hiT[i] = (_Float16*)alloc(256 * 256 * 2);
    gA2T[i]   = (_Float16*)alloc(64 * 256 * 2);
    gW1T[i]   = (_Float16*)alloc((size_t)1024 * J17C * 2);
    gW2T[i]   = (_Float16*)alloc((size_t)256 * 1024 * 2);
  }

  auto wconv = [&](const float* W, _Float16* Wt, int Kin, int Kpad, int Nin) {
    size_t tot = (size_t)Nin * Kpad;
    k_wconv<<<dim3((unsigned)((tot + 255) / 256)), dim3(256), 0, stream>>>(W, Wt, Kin, Kpad, Nin);
  };
  auto gemmRF16 = [&](const _Float16* A, const _Float16* Bt, const float* bias,
                      _Float16* out, int M, int K, int N) {
    k_gemm<true, true, false><<<dim3(N / 64, M / 64), dim3(128), 0, stream>>>(
        A, Bt, bias, nullptr, nullptr, (void*)out, M, K, N);
  };
  auto gemmRF32 = [&](const _Float16* A, const _Float16* Bt, const float* bias,
                      float* out, int M, int K, int N) {
    k_gemm<true, false, false><<<dim3(N / 64, M / 64), dim3(128), 0, stream>>>(
        A, Bt, bias, nullptr, nullptr, (void*)out, M, K, N);
  };
  auto gemmF32 = [&](const _Float16* A, const _Float16* Bt, float* out, int M, int K, int N) {
    k_gemm<false, false, false><<<dim3(N / 64, M / 64), dim3(128), 0, stream>>>(
        A, Bt, nullptr, nullptr, nullptr, (void*)out, M, K, N);
  };
  auto gemmPQ = [&](const _Float16* A, const _Float16* Bt, const float* bias,
                    const float* P, const float* Q, _Float16* out, int M, int K, int N) {
    k_gemm<true, true, true><<<dim3(N / 64, M / 64), dim3(128), 0, stream>>>(
        A, Bt, bias, P, Q, (void*)out, M, K, N);
  };

  // ---- weight conversion (transpose to [N x Kpad] f16) ----
  wconv(We1, We1T, 34, 64, 256);
  wconv(We2, We2T, 256, 256, 256);
  for (int i = 0; i < 2; ++i) {
    wconv(vA1 + (size_t)i * 256 * 256, vA1T[i], 256, 256, 256);
    wconv(vA2 + (size_t)i * 256 * 64,  vA2T[i], 256, 256, 64);
    wconv(gA1 + ((size_t)i * 768 + 0)   * 256, gA1loT[i],  256, 256, 256);
    wconv(gA1 + ((size_t)i * 768 + 256) * 256, gA1midT[i], 256, 256, 256);
    wconv(gA1 + ((size_t)i * 768 + 512) * 256, gA1hiT[i],  256, 256, 256);
    wconv(gA2 + (size_t)i * 256 * 64,  gA2T[i], 256, 256, 64);
    wconv(gW1 + (size_t)i * J17C * 1024, gW1T[i], J17C, J17C, 1024);
    wconv(gW2 + (size_t)i * 1024 * 256,  gW2T[i], 1024, 1024, 256);
  }

  // ---- node embedding ----
  k_node_embed<<<dim3(NPAD), dim3(256), 0, stream>>>(vis, Wv, bv, Wvf, bvf, jt, nf);

  // ---- geometric edge features ef = relu(relu(x_ij@We1+be1)@We2+be2) ----
  k_build_xa<<<dim3(NPAIR * 64 / 256), dim3(256), 0, stream>>>(xy, B64);
  gemmRF16(B64, We1T, be1, BIGA, NPAIR, 64, 256);
  gemmRF16(BIGA, We2T, be2, BIGB, NPAIR, 256, 256);   // ef (f16, persists)

  // ---- visual iterations ----
  const float* nfc = nf;
  for (int i = 0; i < 2; ++i) {
    k_build_da<<<dim3(NPAIR), dim3(256), 0, stream>>>(nfc, BIGC);
    gemmRF16(BIGC, vA1T[i], vbA1 + i * 256, BIGA, NPAIR, 256, 256);
    gemmRF16(BIGA, vA2T[i], vbA2 + i * 64, B64, NPAIR, 256, 64);
    k_mlp_last<<<dim3(NPAIR / 256), dim3(256), 0, stream>>>(B64, vA3 + i * 64, vbA3 + i, evis);
    if (i == 0) {
      k_vis_update<<<dim3(NPAD), dim3(256), 0, stream>>>(evis, nf, vWn, vbn, nf2);
      nfc = nf2;
    }
  }

  // ---- geometric iterations ----
  k_norm_joint<<<dim3(N_NODE), dim3(256), 0, stream>>>(edge, N_NODE, 0, eg);
  for (int i = 0; i < 2; ++i) {
    k_segmean<<<dim3(NPAD, 17), dim3(256), 0, stream>>>(eg, BIGB, normf);
    gemmRF16(normf, gW1T[i], gb1 + i * 1024, t1, NPAD, J17C, 1024);
    gemmRF32(t1, gW2T[i], gb2 + i * 256, aggf, NPAD, 1024, 256);
    k_nfe<<<dim3(NPAD), dim3(256), 0, stream>>>(aggf, gWn + (size_t)i * J17C * 256,
                                                gbn + i * 256, jt, nfeh);
    // rank decomposition of cat@gA1: P = nfe@gA1_mid, Q = nfe@gA1_hi
    gemmF32(nfeh, gA1midT[i], Pbuf, NPAD, 256, 256);
    gemmF32(nfeh, gA1hiT[i],  Qbuf, NPAD, 256, 256);
    gemmPQ(BIGB, gA1loT[i], gbA1 + i * 256, Pbuf, Qbuf, BIGC, NPAIR, 256, 256);
    gemmRF16(BIGC, gA2T[i], gbA2 + i * 64, B64, NPAIR, 256, 64);
    k_mlp_last<<<dim3(NPAIR / 256), dim3(256), 0, stream>>>(B64, gA3 + i * 64, gbA3 + i, sim);
    if (i == 0) k_norm_joint<<<dim3(N_NODE), dim3(256), 0, stream>>>(sim, NPAD, 1, eg);
  }

  // ---- merge geometric + visual affinities ----
  k_merge<<<dim3(N_NODE), dim3(256), 0, stream>>>(sim, evis, mW1, mb1, mW2, mb2, mW3, mb3,
                                                  mW4, mb4, mW5, mb5, (float*)d_out);
}
